// CrossAttentionBlock_53188874994142
// MI455X (gfx1250) — matrix operand order repaired and checked by reading, not compiled, not run on hardware
//
#include <hip/hip_runtime.h>
#include <hip/hip_bf16.h>

// ---------------------------------------------------------------------------
// CDNA5 (gfx1250) cross-attention block, bf16 WMMA path.
// D = A x B^T everywhere: B operand is N x K row-major, which matches the
// documented V_WMMA B-fragment layout (lane = column, K contiguous per lane).
// Each wave computes a 64x32 output tile (4x2 WMMA tiles): 8 WMMAs per
// 6 fragment loads per K-step (~21 FLOP/B from cache).
// Final (never-re-read) f32 outputs use non-temporal stores so the L2 keeps
// the scores/attn intermediates instead.
// ---------------------------------------------------------------------------

typedef __attribute__((ext_vector_type(16))) __bf16 v16bf;
typedef __attribute__((ext_vector_type(8)))  float  v8f;

#define DEV __device__ __forceinline__

static constexpr int Bn   = 32;     // batch
static constexpr int P    = 576;    // image patch tokens
static constexpr int Qt   = 256;    // tactile tokens
static constexpr int D    = 768;    // model dim
static constexpr int H    = 8;      // heads
static constexpr int DH   = 96;     // head dim
static constexpr int LIMG = 577;    // 1 + P

union ABPack { uint4 q[2]; v16bf v; };

DEV void st_nt(float* p, float v) { __builtin_nontemporal_store(v, p); }

// A fragment (16x32 bf16, row-major, stride lda):
//  lanes 0-15 : M=lane,    K = k0+0..7   (elems 0..7),  k0+16..23 (elems 8..15)
//  lanes16-31 : M=lane-16, K = k0+8..15,               k0+24..31
DEV v16bf load_a_frag(const __bf16* A, int lda, int mbase, int k0) {
  int lane = threadIdx.x & 31;
  int row  = mbase + (lane & 15);
  int ka   = k0 + ((lane >> 4) << 3);
  const __bf16* p = A + (long)row * lda + ka;
  ABPack t;
  t.q[0] = *reinterpret_cast<const uint4*>(p);
  t.q[1] = *reinterpret_cast<const uint4*>(p + 16);
  return t.v;
}

// B fragment (32x16 bf16 in K x N sense; stored as N x K row-major, stride ldb):
//  lanes 0-15 : N=lane,    K = k0+0..7   (elems 0..7),  k0+16..23 (elems 8..15)
//  lanes16-31 : N=lane-16, K = k0+8..15,               k0+24..31
DEV v16bf load_b_frag(const __bf16* Bm, int ldb, int nbase, int k0) {
  int lane = threadIdx.x & 31;
  int col  = nbase + (lane & 15);
  int kb   = k0 + ((lane >> 4) << 3);
  const __bf16* p = Bm + (long)col * ldb + kb;
  ABPack t;
  t.q[0] = *reinterpret_cast<const uint4*>(p);
  t.q[1] = *reinterpret_cast<const uint4*>(p + 16);
  return t.v;
}

// Store modes:
//  0: scores      f32out[z*cBatch + m*ldc + n]              (batched, no bias)
//  1: split-head  bfout[((b*H+h)*Lseq + q)*DH + d] = acc + bias[n]
//  2: split-headT bfout[((b*H+h)*DH + d)*Lseq + q] = acc + bias[n]   (V^T)
//  3: context     bfout[(b*Lseq + m)*D + h*DH + n]          (z = b*H + h)
//  4: out-proj    f32out[m*D + n] = acc + bias[n] + residual
template <int MODE>
DEV void store_tile(v8f acc, int mt, int nt, int z,
                    float* f32out, long cBatch, int ldc, __bf16* bfout,
                    const float* bias, const float* res, int resMode, int Lseq) {
  float bv = 0.f;
  if (MODE == 1 || MODE == 2 || MODE == 4) bv = bias[nt];
  int h = 0, d = 0;
  if (MODE == 1 || MODE == 2) { h = nt / DH; d = nt - h * DH; }
#pragma unroll
  for (int r = 0; r < 8; ++r) {
    int   m = mt + r;
    float v = acc[r];
    if (MODE == 0) {
      f32out[(long)z * cBatch + (long)m * ldc + nt] = v;
    } else if (MODE == 1) {
      int bb = m / Lseq, q = m - bb * Lseq;
      bfout[((long)((bb * H + h) * Lseq + q)) * DH + d] = (__bf16)(v + bv);
    } else if (MODE == 2) {
      int bb = m / Lseq, q = m - bb * Lseq;
      bfout[((long)((bb * H + h) * DH + d)) * Lseq + q] = (__bf16)(v + bv);
    } else if (MODE == 3) {
      int bb = z >> 3, hh = z & 7;
      bfout[((long)(bb * Lseq + m)) * D + hh * DH + nt] = (__bf16)v;
    } else { // MODE == 4
      float rv;
      if (resMode == 0) {
        rv = res[(long)m * D + nt];
      } else { // image rows: skip cls token per batch
        int bb = m / P, pp = m - bb * P;
        rv = res[((long)bb * LIMG + 1 + pp) * D + nt];
      }
      f32out[(long)m * D + nt] = v + bv + rv;
    }
  }
}

// One wave computes a 64x32 output tile (4x2 WMMA tiles); 8 waves / block.
template <int MODE>
__global__ __launch_bounds__(256) void wmma_gemm_k(
    const __bf16* __restrict__ A, long aBatch, int lda,
    const __bf16* __restrict__ Bm, long bBatch, int ldb,
    int M, int N, int Kd,
    float* __restrict__ f32out, long cBatch, int ldc,
    __bf16* __restrict__ bfout,
    const float* __restrict__ bias, const float* __restrict__ res,
    int resMode, int Lseq) {
  int z = blockIdx.z;
  const __bf16* Az = A  + (long)z * aBatch;
  const __bf16* Bz = Bm + (long)z * bBatch;

  int tilesN = N >> 5, tilesM = M >> 6;
  int gw = blockIdx.x * 8 + (threadIdx.x >> 5);
  if (gw >= tilesM * tilesN) return;
  int tm = gw / tilesN, tn = gw - tm * tilesN;
  int mbase = tm << 6, nbase = tn << 5;

  v8f acc[8] = {{}, {}, {}, {}, {}, {}, {}, {}};  // acc[i*2+j]
  for (int k0 = 0; k0 < Kd; k0 += 32) {
    v16bf a[4], b[2];
#pragma unroll
    for (int i = 0; i < 4; ++i) a[i] = load_a_frag(Az, lda, mbase + 16 * i, k0);
#pragma unroll
    for (int j = 0; j < 2; ++j) b[j] = load_b_frag(Bz, ldb, nbase + 16 * j, k0);
#pragma unroll
    for (int i = 0; i < 4; ++i)
#pragma unroll
      for (int j = 0; j < 2; ++j)
        acc[i * 2 + j] = __builtin_amdgcn_wmma_f32_16x16x32_bf16(
            false, a[i], false, b[j], (short)0, acc[i * 2 + j], false, false);
  }

  int lane = threadIdx.x & 31;
  int mo = (lane >> 4) << 3;   // C/D layout: M = r + 8*(lane>>4)
  int nl = lane & 15;          //             N = lane & 15
#pragma unroll
  for (int i = 0; i < 4; ++i)
#pragma unroll
    for (int j = 0; j < 2; ++j)
      store_tile<MODE>(acc[i * 2 + j], mbase + 16 * i + mo, nbase + 16 * j + nl,
                       z, f32out, cBatch, ldc, bfout, bias, res, resMode, Lseq);
}

// Row softmax with 1/sqrt(dh) scaling; optional f32 output (attn weights,
// streamed with NT stores) + bf16 copy feeding the P*V WMMA GEMM.
__global__ __launch_bounds__(256) void softmax_k(
    const float* __restrict__ S, float* __restrict__ outF,
    __bf16* __restrict__ outB, int L, float scale) {
  __shared__ float red[256];
  long row = blockIdx.x;
  const float* x = S + row * L;
  int tid = threadIdx.x;

  float m = -1e30f;
  for (int i = tid; i < L; i += 256) m = fmaxf(m, x[i]);
  red[tid] = m; __syncthreads();
  for (int s = 128; s > 0; s >>= 1) { if (tid < s) red[tid] = fmaxf(red[tid], red[tid + s]); __syncthreads(); }
  m = red[0]; __syncthreads();

  float sum = 0.f;
  for (int i = tid; i < L; i += 256) sum += __expf((x[i] - m) * scale);
  red[tid] = sum; __syncthreads();
  for (int s = 128; s > 0; s >>= 1) { if (tid < s) red[tid] += red[tid + s]; __syncthreads(); }
  float inv = 1.f / red[0];

  for (int i = tid; i < L; i += 256) {
    float p = __expf((x[i] - m) * scale) * inv;
    if (outF) st_nt(outF + row * L + i, p);
    outB[row * L + i] = (__bf16)p;
  }
}

// LayerNorm over D=768; imgMap remaps output rows past the cls token.
// f32 output (final, never re-read on device) uses NT stores.
__global__ __launch_bounds__(256) void ln_k(
    const float* __restrict__ X, const float* __restrict__ g,
    const float* __restrict__ bt, float* __restrict__ outF,
    __bf16* __restrict__ outB, int imgMap) {
  __shared__ float r1[256], r2[256];
  int row = blockIdx.x, tid = threadIdx.x;
  const float* x = X + (long)row * D;
  float s = 0.f, s2 = 0.f;
  for (int i = tid; i < D; i += 256) { float v = x[i]; s += v; s2 += v * v; }
  r1[tid] = s; r2[tid] = s2; __syncthreads();
  for (int st = 128; st > 0; st >>= 1) {
    if (tid < st) { r1[tid] += r1[tid + st]; r2[tid] += r2[tid + st]; }
    __syncthreads();
  }
  float mean = r1[0] * (1.f / D);
  float var  = r2[0] * (1.f / D) - mean * mean;
  float rstd = rsqrtf(var + 1e-5f);
  long orow = row;
  if (imgMap) { int bb = row / P, pp = row - bb * P; orow = (long)bb * LIMG + 1 + pp; }
  for (int i = tid; i < D; i += 256) {
    float y = (x[i] - mean) * rstd * g[i] + bt[i];
    st_nt(outF + orow * D + i, y);
    if (outB) outB[(long)row * D + i] = (__bf16)y;
  }
}

__global__ void cvt_k(const float* __restrict__ s, __bf16* __restrict__ d, long n) {
  long i = (long)blockIdx.x * 256 + threadIdx.x;
  if (i < n) d[i] = (__bf16)s[i];
}

__global__ void cvt_patches_k(const float* __restrict__ img, __bf16* __restrict__ d) {
  long i = (long)blockIdx.x * 256 + threadIdx.x;
  const long n = (long)Bn * P * D;
  if (i < n) {
    long b = i / ((long)P * D);
    long r = i - b * (long)P * D;
    d[i] = (__bf16)img[b * (long)LIMG * D + D + r];
  }
}

__global__ void cls_copy_k(const float* __restrict__ img, float* __restrict__ out) {
  int b = blockIdx.x;
  for (int i = threadIdx.x; i < D; i += 256) {
    long o = (long)b * LIMG * D + i;
    st_nt(out + o, img[o]);
  }
}

static void launch_gemm(int mode, hipStream_t st,
                        const __bf16* A, long aB, int lda,
                        const __bf16* Bm, long bB, int ldb,
                        int M, int N, int K, int batches,
                        float* f32o, long cB, int ldc, __bf16* bfo,
                        const float* bias, const float* res, int resMode, int Lseq) {
  int tiles = (M / 64) * (N / 32);
  dim3 grid((unsigned)((tiles + 7) / 8), 1, (unsigned)batches);
  dim3 blk(256, 1, 1);
  switch (mode) {
  case 0: wmma_gemm_k<0><<<grid, blk, 0, st>>>(A, aB, lda, Bm, bB, ldb, M, N, K, f32o, cB, ldc, bfo, bias, res, resMode, Lseq); break;
  case 1: wmma_gemm_k<1><<<grid, blk, 0, st>>>(A, aB, lda, Bm, bB, ldb, M, N, K, f32o, cB, ldc, bfo, bias, res, resMode, Lseq); break;
  case 2: wmma_gemm_k<2><<<grid, blk, 0, st>>>(A, aB, lda, Bm, bB, ldb, M, N, K, f32o, cB, ldc, bfo, bias, res, resMode, Lseq); break;
  case 3: wmma_gemm_k<3><<<grid, blk, 0, st>>>(A, aB, lda, Bm, bB, ldb, M, N, K, f32o, cB, ldc, bfo, bias, res, resMode, Lseq); break;
  default: wmma_gemm_k<4><<<grid, blk, 0, st>>>(A, aB, lda, Bm, bB, ldb, M, N, K, f32o, cB, ldc, bfo, bias, res, resMode, Lseq); break;
  }
}

extern "C" void kernel_launch(void* const* d_in, const int* in_sizes, int n_in,
                              void* d_out, int out_size, void* d_ws, size_t ws_size,
                              hipStream_t stream) {
  const float* img      = (const float*)d_in[0];
  const float* tact     = (const float*)d_in[1];
  const float* t2i_in_w = (const float*)d_in[2];
  const float* t2i_in_b = (const float*)d_in[3];
  const float* t2i_ow   = (const float*)d_in[4];
  const float* t2i_ob   = (const float*)d_in[5];
  const float* i2t_in_w = (const float*)d_in[6];
  const float* i2t_in_b = (const float*)d_in[7];
  const float* i2t_ow   = (const float*)d_in[8];
  const float* i2t_ob   = (const float*)d_in[9];
  const float* g_t = (const float*)d_in[10];
  const float* b_t = (const float*)d_in[11];
  const float* g_i = (const float*)d_in[12];
  const float* b_i = (const float*)d_in[13];
  float* out = (float*)d_out;

  const long nImg   = (long)Bn * P * D;       // 14,155,776
  const long nTact  = (long)Bn * Qt * D;      //  6,291,456
  const long nInW   = (long)3 * D * D;        //  1,769,472
  const long nOutW  = (long)D * D;            //    589,824
  const long nAttn  = (long)Bn * H * Qt * P;  // 37,748,736

  const long outImgOff  = 0;
  const long outTactOff = (long)Bn * LIMG * D;         // 14,180,352
  const long outAttnOff = outTactOff + nTact;          // 20,471,808

  char* w = (char*)d_ws;
  size_t off = 0;
  auto alloc = [&](size_t bytes) -> void* {
    void* p = w + off;
    off += (bytes + 255) & ~(size_t)255;
    return p;
  };
  __bf16* xImg    = (__bf16*)alloc(nImg * 2);
  __bf16* xTact   = (__bf16*)alloc(nTact * 2);
  __bf16* wT2iIn  = (__bf16*)alloc(nInW * 2);
  __bf16* wT2iOut = (__bf16*)alloc(nOutW * 2);
  __bf16* wI2tIn  = (__bf16*)alloc(nInW * 2);
  __bf16* wI2tOut = (__bf16*)alloc(nOutW * 2);
  __bf16* Qp      = (__bf16*)alloc(nImg * 2);   // sized for the larger layer
  __bf16* Kp      = (__bf16*)alloc(nImg * 2);
  __bf16* Vt      = (__bf16*)alloc(nImg * 2);
  float*  scores  = (float*)alloc(nAttn * 4);
  __bf16* attnB   = (__bf16*)alloc(nAttn * 2);
  __bf16* ctxB    = (__bf16*)alloc(nImg * 2);
  float*  yPre    = (float*)alloc(nImg * 4);
  __bf16* tactNew = (__bf16*)alloc(nTact * 2);
  (void)ws_size; (void)in_sizes; (void)n_in; (void)out_size;

  const float scale = 0.10206207261596577f; // 1/sqrt(96)

  // ---- f32 -> bf16 conversions -------------------------------------------
  cvt_patches_k<<<(unsigned)((nImg + 255) / 256), 256, 0, stream>>>(img, xImg);
  cvt_k<<<(unsigned)((nTact + 255) / 256), 256, 0, stream>>>(tact, xTact, nTact);
  cvt_k<<<(unsigned)((nInW  + 255) / 256), 256, 0, stream>>>(t2i_in_w, wT2iIn,  nInW);
  cvt_k<<<(unsigned)((nOutW + 255) / 256), 256, 0, stream>>>(t2i_ow,   wT2iOut, nOutW);
  cvt_k<<<(unsigned)((nInW  + 255) / 256), 256, 0, stream>>>(i2t_in_w, wI2tIn,  nInW);
  cvt_k<<<(unsigned)((nOutW + 255) / 256), 256, 0, stream>>>(i2t_ow,   wI2tOut, nOutW);

  // ======================= MHA 1: tactile -> image ========================
  // Q = tact @ Wq^T + bq  (split-head, (B,H,Qt,DH))
  launch_gemm(1, stream, xTact, 0, D, wT2iIn, 0, D, Bn * Qt, D, D, 1,
              nullptr, 0, 0, Qp, t2i_in_b, nullptr, 0, Qt);
  // K = patches @ Wk^T + bk
  launch_gemm(1, stream, xImg, 0, D, wT2iIn + (long)D * D, 0, D, Bn * P, D, D, 1,
              nullptr, 0, 0, Kp, t2i_in_b + D, nullptr, 0, P);
  // V^T = (patches @ Wv^T + bv)^T per head  ((B,H,DH,P))
  launch_gemm(2, stream, xImg, 0, D, wT2iIn + 2L * D * D, 0, D, Bn * P, D, D, 1,
              nullptr, 0, 0, Vt, t2i_in_b + 2 * D, nullptr, 0, P);
  // scores = Q K^T  (batched over B*H, M=Qt, N=P, K=DH)
  launch_gemm(0, stream, Qp, (long)Qt * DH, DH, Kp, (long)P * DH, DH,
              Qt, P, DH, Bn * H, scores, (long)Qt * P, P, nullptr,
              nullptr, nullptr, 0, 0);
  // softmax -> f32 attn output (NT) + bf16 copy
  softmax_k<<<(unsigned)(Bn * H * Qt), 256, 0, stream>>>(
      scores, out + outAttnOff, attnB, P, scale);
  // ctx = attn @ V  -> (B, Qt, D) bf16
  launch_gemm(3, stream, attnB, (long)Qt * P, P, Vt, (long)DH * P, P,
              Qt, DH, P, Bn * H, nullptr, 0, 0, ctxB,
              nullptr, nullptr, 0, Qt);
  // out-proj + bias + residual(tact)
  launch_gemm(4, stream, ctxB, 0, D, wT2iOut, 0, D, Bn * Qt, D, D, 1,
              yPre, 0, D, nullptr, t2i_ob, tact, 0, 0);
  // LayerNorm -> f32 tactile output + bf16 for MHA2 K/V
  ln_k<<<(unsigned)(Bn * Qt), 256, 0, stream>>>(
      yPre, g_t, b_t, out + outTactOff, tactNew, 0);

  // ======================= MHA 2: image -> tactile ========================
  launch_gemm(1, stream, xImg, 0, D, wI2tIn, 0, D, Bn * P, D, D, 1,
              nullptr, 0, 0, Qp, i2t_in_b, nullptr, 0, P);
  launch_gemm(1, stream, tactNew, 0, D, wI2tIn + (long)D * D, 0, D, Bn * Qt, D, D, 1,
              nullptr, 0, 0, Kp, i2t_in_b + D, nullptr, 0, Qt);
  launch_gemm(2, stream, tactNew, 0, D, wI2tIn + 2L * D * D, 0, D, Bn * Qt, D, D, 1,
              nullptr, 0, 0, Vt, i2t_in_b + 2 * D, nullptr, 0, Qt);
  launch_gemm(0, stream, Qp, (long)P * DH, DH, Kp, (long)Qt * DH, DH,
              P, Qt, DH, Bn * H, scores, (long)P * Qt, Qt, nullptr,
              nullptr, nullptr, 0, 0);
  softmax_k<<<(unsigned)(Bn * H * P), 256, 0, stream>>>(
      scores, nullptr, attnB, Qt, scale);
  launch_gemm(3, stream, attnB, (long)P * Qt, Qt, Vt, (long)DH * Qt, Qt,
              P, DH, Qt, Bn * H, nullptr, 0, 0, ctxB,
              nullptr, nullptr, 0, P);
  launch_gemm(4, stream, ctxB, 0, D, wI2tOut, 0, D, Bn * P, D, D, 1,
              yPre, 0, D, nullptr, i2t_ob, img, 1, 0);
  // LayerNorm -> f32 image output rows (cls rows filled separately)
  ln_k<<<(unsigned)(Bn * P), 256, 0, stream>>>(
      yPre, g_i, b_i, out + outImgOff, nullptr, 1);
  cls_copy_k<<<Bn, 256, 0, stream>>>(img, out + outImgOff);
}
